// MetaLSTM_86423331930502
// MI455X (gfx1250) — matrix operand order repaired and checked by reading, not compiled, not run on hardware
//
#include <hip/hip_runtime.h>
#include <hip/hip_bf16.h>

// ---------------- types ----------------
typedef __bf16 bf16_t;
typedef __attribute__((ext_vector_type(16))) __bf16 v16bf;
typedef __attribute__((ext_vector_type(8)))  float  v8f;
typedef __attribute__((ext_vector_type(4)))  float  f32x4;

#define T_DIM 128
#define B_DIM 512
#define DIN_DIM 512
#define H_DIM 512
#define DM_DIM 256
#define HYP_DIM 256
#define L_DIM 100
#define LP_DIM 128   // L padded to 128 for WMMA tiling

#define GEMM_RELU    1
#define GEMM_OUTBF16 2

// float -> bf16 (round-to-nearest-even) without relying on __bf16 conversions
__device__ __forceinline__ bf16_t tobf(float f) {
  union { float f; unsigned u; } x; x.f = f;
  unsigned r = x.u + 0x7FFFu + ((x.u >> 16) & 1u);
  unsigned short s = (unsigned short)(r >> 16);
  union { unsigned short s; bf16_t b; } y; y.s = s; return y.b;
}

union FragU { v16bf v; f32x4 f[2]; };

// A-matrix 16x32 bf16 fragment (rows contiguous in K, row stride = lda elems)
// lanes 0-15: row=lane, K = [0..7] then [16..23]; lanes 16-31: row=lane-16, K=[8..15],[24..31]
__device__ __forceinline__ v16bf load_frag_a(const bf16_t* tile, long lda, int lane) {
  const bool lo = lane < 16;
  const bf16_t* p = tile + (long)(lane & 15) * lda + (lo ? 0 : 8);
  FragU u;
  u.f[0] = *(const f32x4*)p;
  u.f[1] = *(const f32x4*)(p + 16);
  return u.v;
}

// B-matrix 32x16 fragment sourced from W (N,K) row-major: B[k][n] = W[n][k]
// lanes 0-15: col=lane, K = [0..7] then [16..23]; lanes 16-31: col=lane-16, K=[8..15],[24..31]
__device__ __forceinline__ v16bf load_frag_b(const bf16_t* tile, long ldb, int lane) {
  const bool lo = lane < 16;
  const bf16_t* p = tile + (long)(lane & 15) * ldb + (lo ? 0 : 8);
  FragU u;
  u.f[0] = *(const f32x4*)p;
  u.f[1] = *(const f32x4*)(p + 16);
  return u.v;
}

// ---------------- generic batched bf16 WMMA GEMM ----------------
// C[m,n] = sum_k A[m,k] * W[n,k]  (+ bias[n]) (+ relu), per batch z offsets.
// Requires M%64==0, N%64==0, K%32==0. Block: 128 thr (4 waves), tile 64x64.
__global__ void __launch_bounds__(128)
gemm_bf16_kernel(const bf16_t* __restrict__ A, long lda, long aBatch,
                 const bf16_t* __restrict__ W, long ldb, long bBatch,
                 void* __restrict__ Cout, long ldc, long cBatch,
                 const float* __restrict__ bias, long biasBatch,
                 int K, int flags)
{
  const int bz   = blockIdx.z;
  A += (long)bz * aBatch;
  W += (long)bz * bBatch;
  const int lane = threadIdx.x & 31;
  const int wave = threadIdx.x >> 5;
  const int m0 = blockIdx.y * 64 + (wave >> 1) * 32;
  const int n0 = blockIdx.x * 64 + (wave & 1) * 32;

  v8f acc[2][2] = {};
  for (int k0 = 0; k0 < K; k0 += 32) {
    v16bf a0 = load_frag_a(A + (long)m0 * lda + k0,        lda, lane);
    v16bf a1 = load_frag_a(A + (long)(m0 + 16) * lda + k0, lda, lane);
    v16bf b0 = load_frag_b(W + (long)n0 * ldb + k0,        ldb, lane);
    v16bf b1 = load_frag_b(W + (long)(n0 + 16) * ldb + k0, ldb, lane);
    acc[0][0] = __builtin_amdgcn_wmma_f32_16x16x32_bf16(false, a0, false, b0, (short)0, acc[0][0], false, false);
    acc[0][1] = __builtin_amdgcn_wmma_f32_16x16x32_bf16(false, a0, false, b1, (short)0, acc[0][1], false, false);
    acc[1][0] = __builtin_amdgcn_wmma_f32_16x16x32_bf16(false, a1, false, b0, (short)0, acc[1][0], false, false);
    acc[1][1] = __builtin_amdgcn_wmma_f32_16x16x32_bf16(false, a1, false, b1, (short)0, acc[1][1], false, false);
  }

  const int col = lane & 15;
  const int rb  = (lane < 16) ? 0 : 8;
  for (int mi = 0; mi < 2; ++mi)
    for (int ni = 0; ni < 2; ++ni) {
      const int nn = n0 + ni * 16 + col;
      const float bv = bias ? bias[(long)bz * biasBatch + nn] : 0.0f;
      for (int r = 0; r < 8; ++r) {
        const int mm = m0 + mi * 16 + rb + r;
        float v = acc[mi][ni][r] + bv;
        if (flags & GEMM_RELU) v = fmaxf(v, 0.0f);
        const long idx = (long)mm * ldc + nn + (long)bz * cBatch;
        if (flags & GEMM_OUTBF16) ((bf16_t*)Cout)[idx] = tobf(v);
        else                      ((float*)Cout)[idx]  = v;
      }
    }
}

// ---------------- attention: scores -> softmax -> attended (fused, WMMA) ----------------
// grid = (T/16, B); 1 wave per block handles 16 t-rows for one batch b.
__global__ void __launch_bounds__(32)
attn_kernel(const bf16_t* __restrict__ qb,   // (T,B,HYP) bf16
            const bf16_t* __restrict__ kb,   // (B,LP,HYP) bf16 (padded rows)
            const bf16_t* __restrict__ mtp,  // (B,DM,LP) bf16 (meta transposed, padded)
            const int*    __restrict__ inp,  // (B,L)
            bf16_t*       __restrict__ att)  // (T,B,DM) bf16
{
  __shared__ __align__(16) bf16_t pTile[16 * LP_DIM];
  const int b    = blockIdx.y;
  const int t0   = blockIdx.x * 16;
  const int lane = threadIdx.x;
  const bool lo  = lane < 16;
  const int col  = lane & 15;
  const int rb   = lo ? 0 : 8;

  // scores (16 x 128) over K = HYP
  v8f s[8] = {};
  const bf16_t* qtile = qb + ((long)t0 * B_DIM + b) * HYP_DIM;
  const long    ldq   = (long)B_DIM * HYP_DIM;
  const bf16_t* ktile = kb + (long)b * LP_DIM * HYP_DIM;
  for (int kt = 0; kt < HYP_DIM / 32; ++kt) {
    v16bf a = load_frag_a(qtile + kt * 32, ldq, lane);
    for (int nt = 0; nt < 8; ++nt) {
      v16bf bf = load_frag_b(ktile + (long)(nt * 16) * HYP_DIM + kt * 32, HYP_DIM, lane);
      s[nt] = __builtin_amdgcn_wmma_f32_16x16x32_bf16(false, a, false, bf, (short)0, s[nt], false, false);
    }
  }

  // scale + mask + row-softmax (rows live across 16-lane halves)
  const float scale = 0.0625f; // 1/sqrt(256)
  float rmax[8];
  for (int r = 0; r < 8; ++r) rmax[r] = -3.0e38f;
  for (int nt = 0; nt < 8; ++nt) {
    const int l = nt * 16 + col;
    const bool valid = (l < L_DIM) && (inp[(long)b * L_DIM + (l < L_DIM ? l : 0)] != 0);
    for (int r = 0; r < 8; ++r) {
      float sc = valid ? s[nt][r] * scale : -1.0e30f;
      s[nt][r] = sc;
      rmax[r] = fmaxf(rmax[r], sc);
    }
  }
  for (int r = 0; r < 8; ++r)
    for (int o = 1; o < 16; o <<= 1) rmax[r] = fmaxf(rmax[r], __shfl_xor(rmax[r], o, 32));
  float rsum[8] = {};
  for (int nt = 0; nt < 8; ++nt)
    for (int r = 0; r < 8; ++r) {
      float p = (s[nt][r] > -1.0e29f) ? __expf(s[nt][r] - rmax[r]) : 0.0f;
      s[nt][r] = p;
      rsum[r] += p;
    }
  for (int r = 0; r < 8; ++r) {
    for (int o = 1; o < 16; o <<= 1) rsum[r] += __shfl_xor(rsum[r], o, 32);
    rsum[r] = 1.0f / fmaxf(rsum[r], 1e-30f);
  }
  for (int nt = 0; nt < 8; ++nt)
    for (int r = 0; r < 8; ++r)
      pTile[(rb + r) * LP_DIM + nt * 16 + col] = tobf(s[nt][r] * rsum[r]);
  __syncthreads();

  // attended (16 x DM) = attn(16 x 128) @ meta[b] via mtp (DM,LP)
  for (int nt = 0; nt < DM_DIM / 16; ++nt) {
    v8f o = {};
    for (int kt = 0; kt < LP_DIM / 32; ++kt) {
      v16bf a  = load_frag_a(pTile + kt * 32, LP_DIM, lane);
      v16bf bf = load_frag_b(mtp + ((long)b * DM_DIM + nt * 16) * LP_DIM + kt * 32, LP_DIM, lane);
      o = __builtin_amdgcn_wmma_f32_16x16x32_bf16(false, a, false, bf, (short)0, o, false, false);
    }
    for (int r = 0; r < 8; ++r) {
      const int t = t0 + rb + r;
      att[((long)t * B_DIM + b) * DM_DIM + nt * 16 + col] = tobf(o[r]);
    }
  }
}

// ---------------- elementwise helpers ----------------
__global__ void f32_to_bf16_kernel(const float* __restrict__ src, bf16_t* __restrict__ dst, long n) {
  long i = (long)blockIdx.x * 256 + threadIdx.x;
  if (i < n) dst[i] = tobf(src[i]);
}
__global__ void copy_f32_kernel(const float* __restrict__ src, float* __restrict__ dst, long n) {
  long i = (long)blockIdx.x * 256 + threadIdx.x;
  if (i < n) dst[i] = src[i];
}
__global__ void zero_f32_kernel(float* __restrict__ dst, long n) {
  long i = (long)blockIdx.x * 256 + threadIdx.x;
  if (i < n) dst[i] = 0.0f;
}
// meta (B,L,DM) -> meta_pad (B,LP,DM) bf16  and  metaT_pad (B,DM,LP) bf16
__global__ void pad_meta_kernel(const float* __restrict__ meta, bf16_t* __restrict__ mp, bf16_t* __restrict__ mtp) {
  long idx = (long)blockIdx.x * 256 + threadIdx.x; // B*LP*DM
  int m = (int)(idx & (DM_DIM - 1));
  long r = idx >> 8;
  int l = (int)(r & (LP_DIM - 1));
  int b = (int)(r >> 7);
  float v = (l < L_DIM) ? meta[((long)b * L_DIM + l) * DM_DIM + m] : 0.0f;
  bf16_t h = tobf(v);
  mp[idx] = h;
  mtp[((long)b * DM_DIM + m) * LP_DIM + l] = h;
}
// cat(B,1280) = [x_t | att_t | h]
__global__ void build_cat_kernel(const bf16_t* __restrict__ xb, const bf16_t* __restrict__ attb,
                                 const bf16_t* __restrict__ hb, bf16_t* __restrict__ cat, int t) {
  long idx = (long)blockIdx.x * 256 + threadIdx.x; // B*1280
  int cix = (int)(idx % 1280);
  int b   = (int)(idx / 1280);
  bf16_t v;
  if (cix < DIN_DIM)              v = xb[((long)t * B_DIM + b) * DIN_DIM + cix];
  else if (cix < DIN_DIM + DM_DIM) v = attb[((long)t * B_DIM + b) * DM_DIM + (cix - DIN_DIM)];
  else                            v = hb[(long)b * H_DIM + (cix - DIN_DIM - DM_DIM)];
  cat[idx] = v;
}
__global__ void init_state_kernel(float* h, float* c, bf16_t* hb) {
  long i = (long)blockIdx.x * 256 + threadIdx.x; // B*H
  h[i] = 0.0f; c[i] = 0.0f; hb[i] = tobf(0.0f);
}
__global__ void finalize_kernel(const float* __restrict__ h, const float* __restrict__ c, float* __restrict__ outp) {
  long i = (long)blockIdx.x * 256 + threadIdx.x; // B*H
  outp[i] = h[i];
  outp[(long)B_DIM * H_DIM + i] = c[i];
  if (i < (long)B_DIM * HYP_DIM) {
    outp[2L * B_DIM * H_DIM + i] = 0.0f;
    outp[2L * B_DIM * H_DIM + (long)B_DIM * HYP_DIM + i] = 0.0f;
  }
}

// ---------------- per-step LayerNorm + LSTM gate fuse ----------------
// dbuf: (B,12,H) = [d_h(4) d_x(4) d_b(4)] (no bdb); whx: (B,8,H) = [wh(4) wx(4)]
__global__ void __launch_bounds__(512)
step_pointwise_kernel(const float* __restrict__ dbuf, const float* __restrict__ whx,
                      const float* __restrict__ bdb, const float* __restrict__ ln_g,
                      const float* __restrict__ ln_b, float* __restrict__ h, float* __restrict__ c,
                      bf16_t* __restrict__ hb, float* __restrict__ out_t) {
  const int b = blockIdx.x;
  const int j = threadIdx.x; // H
  __shared__ float red_s[16];
  __shared__ float red_q[16];
  const int wid = threadIdx.x >> 5, lid = threadIdx.x & 31;
  float y[4];
  for (int k = 0; k < 4; ++k) {
    const long db_base = (long)b * 12 * H_DIM;
    const long wx_base = (long)b * 8 * H_DIM;
    float dh = dbuf[db_base + (long)k * H_DIM + j];
    float dx = dbuf[db_base + (long)(4 + k) * H_DIM + j];
    float dB = dbuf[db_base + (long)(8 + k) * H_DIM + j] + bdb[(long)k * H_DIM + j];
    float wh = whx[wx_base + (long)k * H_DIM + j];
    float wx = whx[wx_base + (long)(4 + k) * H_DIM + j];
    float v = dh * wh + dx * wx + dB;
    float s1 = v, s2 = v * v;
    for (int o = 16; o >= 1; o >>= 1) { s1 += __shfl_xor(s1, o, 32); s2 += __shfl_xor(s2, o, 32); }
    if (lid == 0) { red_s[wid] = s1; red_q[wid] = s2; }
    __syncthreads();
    float S = 0.0f, Q = 0.0f;
    for (int w = 0; w < 16; ++w) { S += red_s[w]; Q += red_q[w]; }
    __syncthreads();
    const float mu = S * (1.0f / H_DIM);
    const float var = Q * (1.0f / H_DIM) - mu * mu;
    y[k] = (v - mu) * rsqrtf(var + 1e-5f) * ln_g[(long)k * H_DIM + j] + ln_b[(long)k * H_DIM + j];
  }
  const float ig = 1.0f / (1.0f + __expf(-y[0]));
  const float fg = 1.0f / (1.0f + __expf(-y[1]));
  const float og = 1.0f / (1.0f + __expf(-y[3]));
  const long idx = (long)b * H_DIM + j;
  const float cn = fg * c[idx] + ig * tanhf(y[2]);
  const float hn = og * tanhf(cn);
  c[idx] = cn; h[idx] = hn; hb[idx] = tobf(hn);
  out_t[idx] = hn;
}

// ---------------- host ----------------
static inline void launch_gemm(hipStream_t stream, const void* A, long lda, long aB,
                               const void* W, long ldb, long bB,
                               void* C, long ldc, long cB,
                               const void* bias, long biasB,
                               int M, int N, int K, int batch, int flags) {
  dim3 g(N / 64, M / 64, batch), blk(128);
  gemm_bf16_kernel<<<g, blk, 0, stream>>>((const bf16_t*)A, lda, aB, (const bf16_t*)W, ldb, bB,
                                          C, ldc, cB, (const float*)bias, biasB, K, flags);
}
static inline void cvt(hipStream_t stream, const float* s, bf16_t* d, long n) {
  f32_to_bf16_kernel<<<dim3((unsigned)((n + 255) / 256)), dim3(256), 0, stream>>>(s, d, n);
}

extern "C" void kernel_launch(void* const* d_in, const int* in_sizes, int n_in,
                              void* d_out, int out_size, void* d_ws, size_t ws_size,
                              hipStream_t stream) {
  const float* x    = (const float*)d_in[0];
  const float* meta = (const float*)d_in[1];
  const int*   inp  = (const int*)d_in[2];
  const float* Wq = (const float*)d_in[3];  const float* bq = (const float*)d_in[4];
  const float* Wk = (const float*)d_in[5];  const float* bk = (const float*)d_in[6];
  const float* Wm = (const float*)d_in[7];  const float* bm = (const float*)d_in[8];
  const float* Wzh = (const float*)d_in[9];  const float* bzh = (const float*)d_in[10];
  const float* Wzx = (const float*)d_in[11]; const float* bzx = (const float*)d_in[12];
  const float* Wzb = (const float*)d_in[13];
  const float* Wdh = (const float*)d_in[14]; const float* Wdx = (const float*)d_in[15];
  const float* Wdb = (const float*)d_in[16]; const float* bdb = (const float*)d_in[17];
  const float* w_h = (const float*)d_in[18]; const float* w_x = (const float*)d_in[19];
  const float* lng = (const float*)d_in[20]; const float* lnb = (const float*)d_in[21];
  (void)in_sizes; (void)n_in; (void)out_size; (void)ws_size;

  // ---- workspace carve (256B aligned) ----
  char* base = (char*)d_ws;
  size_t off = 0;
  auto alloc = [&](size_t bytes) -> void* {
    void* p = base + off;
    off += (bytes + 255) & ~(size_t)255;
    return p;
  };
  const long nTB = (long)T_DIM * B_DIM;
  bf16_t* xb   = (bf16_t*)alloc(nTB * DIN_DIM * 2);           // x bf16
  bf16_t* qb   = (bf16_t*)alloc(nTB * HYP_DIM * 2);           // q bf16
  bf16_t* attb = (bf16_t*)alloc(nTB * DM_DIM * 2);            // attended bf16
  bf16_t* mpad = (bf16_t*)alloc((long)B_DIM * LP_DIM * DM_DIM * 2);
  bf16_t* kpad = (bf16_t*)alloc((long)B_DIM * LP_DIM * HYP_DIM * 2);
  bf16_t* mtp  = (bf16_t*)alloc((long)B_DIM * DM_DIM * LP_DIM * 2);
  bf16_t* wqb  = (bf16_t*)alloc((long)HYP_DIM * DIN_DIM * 2);
  bf16_t* wkb  = (bf16_t*)alloc((long)HYP_DIM * DM_DIM * 2);
  bf16_t* wmb  = (bf16_t*)alloc((long)HYP_DIM * 1280 * 2);
  bf16_t* wz   = (bf16_t*)alloc((long)3072 * HYP_DIM * 2);    // [Wzh;Wzx;Wzb]
  bf16_t* wd   = (bf16_t*)alloc((long)12 * H_DIM * HYP_DIM * 2); // [Wdh;Wdx;Wdb]
  bf16_t* whxw = (bf16_t*)alloc((long)8 * H_DIM * H_DIM * 2);    // [w_h;w_x]
  float*  bz   = (float*)alloc(3072 * 4);                     // [bzh;bzx;0]
  float*  hf   = (float*)alloc((long)B_DIM * H_DIM * 4);
  float*  cf   = (float*)alloc((long)B_DIM * H_DIM * 4);
  bf16_t* hb   = (bf16_t*)alloc((long)B_DIM * H_DIM * 2);
  bf16_t* cat  = (bf16_t*)alloc((long)B_DIM * 1280 * 2);
  bf16_t* mg   = (bf16_t*)alloc((long)B_DIM * HYP_DIM * 2);
  bf16_t* zb   = (bf16_t*)alloc((long)B_DIM * 3072 * 2);
  float*  dbuf = (float*)alloc((long)B_DIM * 12 * H_DIM * 4);
  float*  whxb = (float*)alloc((long)B_DIM * 8 * H_DIM * 4);

  // ---- weight conversions / concatenations ----
  cvt(stream, Wq, wqb, (long)HYP_DIM * DIN_DIM);
  cvt(stream, Wk, wkb, (long)HYP_DIM * DM_DIM);
  cvt(stream, Wm, wmb, (long)HYP_DIM * 1280);
  cvt(stream, Wzh, wz,                 (long)1024 * HYP_DIM);
  cvt(stream, Wzx, wz + 1024L * HYP_DIM, (long)1024 * HYP_DIM);
  cvt(stream, Wzb, wz + 2048L * HYP_DIM, (long)1024 * HYP_DIM);
  cvt(stream, Wdh, wd,                       4L * H_DIM * HYP_DIM);
  cvt(stream, Wdx, wd + 4L * H_DIM * HYP_DIM, 4L * H_DIM * HYP_DIM);
  cvt(stream, Wdb, wd + 8L * H_DIM * HYP_DIM, 4L * H_DIM * HYP_DIM);
  cvt(stream, w_h, whxw,                     4L * H_DIM * H_DIM);
  cvt(stream, w_x, whxw + 4L * H_DIM * H_DIM, 4L * H_DIM * H_DIM);
  copy_f32_kernel<<<dim3(4), dim3(256), 0, stream>>>(bzh, bz, 1024);
  copy_f32_kernel<<<dim3(4), dim3(256), 0, stream>>>(bzx, bz + 1024, 1024);
  zero_f32_kernel<<<dim3(4), dim3(256), 0, stream>>>(bz + 2048, 1024);

  // ---- activations ----
  cvt(stream, x, xb, nTB * DIN_DIM);
  pad_meta_kernel<<<dim3((unsigned)((long)B_DIM * LP_DIM * DM_DIM / 256)), dim3(256), 0, stream>>>(meta, mpad, mtp);

  // ---- attention phase ----
  // q = x @ Wq^T + bq   (M=T*B, N=HYP, K=DIN) -> bf16
  launch_gemm(stream, xb, DIN_DIM, 0, wqb, DIN_DIM, 0, qb, HYP_DIM, 0, bq, 0,
              (int)nTB, HYP_DIM, DIN_DIM, 1, GEMM_OUTBF16);
  // k = meta_pad @ Wk^T + bk (M=B*LP, N=HYP, K=DM) -> bf16
  launch_gemm(stream, mpad, DM_DIM, 0, wkb, DM_DIM, 0, kpad, HYP_DIM, 0, bk, 0,
              B_DIM * LP_DIM, HYP_DIM, DM_DIM, 1, GEMM_OUTBF16);
  // fused scores/softmax/attended
  attn_kernel<<<dim3(T_DIM / 16, B_DIM), dim3(32), 0, stream>>>(qb, kpad, mtp, inp, attb);

  // ---- recurrence ----
  init_state_kernel<<<dim3((long)B_DIM * H_DIM / 256), dim3(256), 0, stream>>>(hf, cf, hb);
  float* out_f = (float*)d_out;
  for (int t = 0; t < T_DIM; ++t) {
    build_cat_kernel<<<dim3((long)B_DIM * 1280 / 256), dim3(256), 0, stream>>>(xb, attb, hb, cat, t);
    // merged = relu(cat @ Wm^T + bm) -> bf16 (M=B, N=HYP, K=1280)
    launch_gemm(stream, cat, 1280, 0, wmb, 1280, 0, mg, HYP_DIM, 0, bm, 0,
                B_DIM, HYP_DIM, 1280, 1, GEMM_RELU | GEMM_OUTBF16);
    // z = merged @ [Wzh;Wzx;Wzb]^T + [bzh;bzx;0] -> bf16 (M=B, N=3072, K=HYP)
    launch_gemm(stream, mg, HYP_DIM, 0, wz, HYP_DIM, 0, zb, 3072, 0, bz, 0,
                B_DIM, 3072, HYP_DIM, 1, GEMM_OUTBF16);
    // d_{h,x,b}[:,k,:] : batch=12, A k-window slides by 256, W batch stride H*HYP
    launch_gemm(stream, zb, 3072, /*aBatch=*/HYP_DIM, wd, HYP_DIM, (long)H_DIM * HYP_DIM,
                dbuf, (long)12 * H_DIM, /*cBatch=*/H_DIM, nullptr, 0,
                B_DIM, H_DIM, HYP_DIM, 12, 0);
    // wh[:,k,:] = h @ w_h[k]^T : batch=4
    launch_gemm(stream, hb, H_DIM, 0, whxw, H_DIM, (long)H_DIM * H_DIM,
                whxb, (long)8 * H_DIM, H_DIM, nullptr, 0,
                B_DIM, H_DIM, H_DIM, 4, 0);
    // wx[:,k,:] = x_t @ w_x[k]^T : batch=4 (C offset +4*H)
    launch_gemm(stream, xb + (long)t * B_DIM * DIN_DIM, DIN_DIM, 0,
                whxw + 4L * H_DIM * H_DIM, H_DIM, (long)H_DIM * H_DIM,
                whxb + 4L * H_DIM, (long)8 * H_DIM, H_DIM, nullptr, 0,
                B_DIM, H_DIM, DIN_DIM, 4, 0);
    // y combine + LN + LSTM gates; writes out[t], updates h/c/hb
    step_pointwise_kernel<<<dim3(B_DIM), dim3(H_DIM), 0, stream>>>(
        dbuf, whxb, bdb, lng, lnb, hf, cf, hb, out_f + (long)t * B_DIM * H_DIM);
  }

  // ---- tail outputs: hT, cT, h_hat x2 ----
  finalize_kernel<<<dim3((long)B_DIM * H_DIM / 256), dim3(256), 0, stream>>>(
      hf, cf, out_f + (long)T_DIM * B_DIM * H_DIM);
}
